// ArbitrageAttention_86981677679392
// MI455X (gfx1250) — matrix operand order repaired and checked by reading, not compiled, not run on hardware
//
#include <hip/hip_runtime.h>
#include <hip/hip_bf16.h>
#include <math.h>

// ---------------- problem constants (from reference) ----------------
#define BB 4
#define SS 2048
#define DD 4096
#define EE 8
#define HH 32
#define HD 128
#define DQ 1024            // D/4
#define NROWS (BB * SS)    // 8192
#define LRATE 1e-3f
#define ALPHA 0.5f
#define TTA_STEPS 3

typedef __bf16 v16bf __attribute__((ext_vector_type(16)));
typedef float  v8f   __attribute__((ext_vector_type(8)));
typedef unsigned short u16;
typedef unsigned int  v4u __attribute__((ext_vector_type(4)));
typedef int           v4i __attribute__((ext_vector_type(4)));
typedef int           v8i __attribute__((ext_vector_type(8)));

union FragBF { uint4 u[2]; v16bf v; };

#if defined(__has_builtin)
#if __has_builtin(__builtin_amdgcn_tensor_load_to_lds) && __has_builtin(__builtin_amdgcn_s_wait_tensorcnt)
#define USE_TDM 1
#endif
#endif
#ifndef USE_TDM
#define USE_TDM 0
#endif

__device__ __forceinline__ u16 f2bf(float f) {
  unsigned int u = __float_as_uint(f);
  unsigned int r = u + 0x7fffu + ((u >> 16) & 1u);   // round-to-nearest-even
  return (u16)(r >> 16);
}

// block-wide sum over 256 threads (8 waves of 32); sh must hold >= 8 floats
__device__ __forceinline__ float block_sum(float v, float* sh) {
  for (int m = 16; m > 0; m >>= 1) v += __shfl_xor(v, m, 32);
  int lane = threadIdx.x & 31, wave = threadIdx.x >> 5;
  if (lane == 0) sh[wave] = v;
  __syncthreads();
  float t = (threadIdx.x < 8) ? sh[threadIdx.x] : 0.f;
  if (wave == 0) {
    for (int m = 4; m > 0; m >>= 1) t += __shfl_xor(t, m, 32);
    if (lane == 0) sh[0] = t;
  }
  __syncthreads();
  float r = sh[0];
  __syncthreads();
  return r;
}

#if USE_TDM
// ---- Tensor Data Mover: 2D tile (rows x 32 bf16), row stride K elems ------
// D# per CDNA5 ISA (group0 128b, group1 256b; groups 2/3
// zero => 2D tensor). data_size=2 bytes; LDS layout = rows contiguous, 64B/row.
// This toolchain exposes the 6-arg builtin (extra int32x8 group, zero-filled).
__device__ __forceinline__ void tdm_load_2d(unsigned lds_off, const u16* gptr,
                                            int K, int rows) {
  unsigned long long ga = (unsigned long long)(uintptr_t)gptr;
  v4u g0;
  g0[0] = 1u;                                              // count=1, user mode
  g0[1] = lds_off;                                         // lds_addr (bytes)
  g0[2] = (unsigned)(ga & 0xffffffffu);                    // global_addr[31:0]
  g0[3] = (unsigned)((ga >> 32) & 0x01ffffffu) | (2u << 30); // [56:32] | type=2
  v8i g1;
  g1[0] = (1 << 16);                                       // data_size = 2B
  g1[1] = (int)(((unsigned)K & 0xffffu) << 16);            // tensor_dim0[15:0]
  g1[2] = (int)((((unsigned)K >> 16) & 0xffffu) |
                (((unsigned)rows & 0xffffu) << 16));       // dim0 hi | dim1 lo
  g1[3] = (int)(((unsigned)rows >> 16) & 0xffffu) | (32 << 16); // dim1 hi | tile_dim0=32
  g1[4] = rows & 0xffff;                                   // tile_dim1 | tile_dim2=0
  g1[5] = K;                                               // dim0_stride[31:0]
  g1[6] = 0;                                               // stride hi | dim1_stride lo
  g1[7] = 0;
  v4i z4 = {0, 0, 0, 0};
  v8i z8 = {0, 0, 0, 0, 0, 0, 0, 0};
  __builtin_amdgcn_tensor_load_to_lds(g0, g1, z4, z4, z8, 0);
}
#endif

// ---------------- weight convert (+optional transpose) to bf16 -------------
__global__ __launch_bounds__(256) void k_cvt_w(const float* __restrict__ W,
                                               u16* __restrict__ Wb,
                                               u16* __restrict__ Wt,
                                               int rows, int cols) {
  long long i = (long long)blockIdx.x * 256 + threadIdx.x;
  if (i >= (long long)rows * cols) return;
  int r = (int)(i / cols), c = (int)(i % cols);
  u16 v = f2bf(W[i]);
  Wb[i] = v;
  if (Wt) Wt[(long long)c * rows + r] = v;
}

__global__ __launch_bounds__(256) void k_cvt_f32_bf16(const float* __restrict__ x,
                                                      u16* __restrict__ y) {
  long long i = (long long)blockIdx.x * 256 + threadIdx.x;
  y[i] = f2bf(x[i]);
}

// ------------- k/v projection: Y[r,c] = dot(X[r,:], W[c,:]), r<32 ----------
__global__ __launch_bounds__(256) void k_kvproj(const float* __restrict__ X,
                                                const float* __restrict__ W,
                                                float* __restrict__ Y) {
  __shared__ float sh[32 * 8];
  int c = blockIdx.x;
  int tid = threadIdx.x, lane = tid & 31, wave = tid >> 5;
  float acc[32];
#pragma unroll
  for (int r = 0; r < 32; ++r) acc[r] = 0.f;
  const float4* wr = (const float4*)(W + (size_t)c * DD);
  for (int k4 = tid; k4 < DD / 4; k4 += 256) {
    float4 w4 = wr[k4];
#pragma unroll
    for (int r = 0; r < 32; ++r) {
      float4 x4 = ((const float4*)(X + (size_t)r * DD))[k4];
      acc[r] += x4.x * w4.x + x4.y * w4.y + x4.z * w4.z + x4.w * w4.w;
    }
  }
#pragma unroll
  for (int r = 0; r < 32; ++r) {
    float v = acc[r];
    for (int m = 16; m > 0; m >>= 1) v += __shfl_xor(v, m, 32);
    if (lane == 0) sh[r * 8 + wave] = v;
  }
  __syncthreads();
  if (tid < 32) {
    float s = 0.f;
#pragma unroll
    for (int w = 0; w < 8; ++w) s += sh[tid * 8 + w];
    Y[(size_t)tid * DD + c] = s;
  }
}

// ------------- kmean_n = l2norm(mean_e engram_k[b,e,:]) --------------------
__global__ __launch_bounds__(256) void k_kmean(const float* __restrict__ ek,
                                               float* __restrict__ km) {
  __shared__ float sh[8];
  int b = blockIdx.x;
  float ssq = 0.f;
  for (int d = threadIdx.x; d < DD; d += 256) {
    float m = 0.f;
#pragma unroll
    for (int e = 0; e < EE; ++e) m += ek[((size_t)b * EE + e) * DD + d];
    m *= (1.f / EE);
    km[(size_t)b * DD + d] = m;
    ssq += m * m;
  }
  float tot = block_sum(ssq, sh);
  float inv = 1.f / fmaxf(sqrtf(tot), 1e-12f);
  for (int d = threadIdx.x; d < DD; d += 256) km[(size_t)b * DD + d] *= inv;
}

// ------------- cross-attention over E=8 + fusion: h = paged + a*out --------
__global__ __launch_bounds__(256) void k_attn_fuse(const float* __restrict__ paged,
                                                   const float* __restrict__ query,
                                                   const float* __restrict__ kp,
                                                   const float* __restrict__ vp,
                                                   float* __restrict__ h) {
  int blk = blockIdx.x;                 // b*S + s
  int b = blk / SS;
  int lane = threadIdx.x & 31, wave = threadIdx.x >> 5;
  const float scale = 0.08838834764831845f;  // 1/sqrt(128)
  for (int hh = wave; hh < HH; hh += 8) {
    size_t qoff = (size_t)blk * DD + hh * HD + lane * 4;
    float4 q4 = *(const float4*)(query + qoff);
    float sc[EE];
#pragma unroll
    for (int e = 0; e < EE; ++e) {
      float4 k4 = *(const float4*)(kp + ((size_t)b * EE + e) * DD + hh * HD + lane * 4);
      float d = q4.x * k4.x + q4.y * k4.y + q4.z * k4.z + q4.w * k4.w;
      for (int m = 16; m > 0; m >>= 1) d += __shfl_xor(d, m, 32);
      sc[e] = d * scale;
    }
    float mx = sc[0];
#pragma unroll
    for (int e = 1; e < EE; ++e) mx = fmaxf(mx, sc[e]);
    float sum = 0.f;
#pragma unroll
    for (int e = 0; e < EE; ++e) { sc[e] = expf(sc[e] - mx); sum += sc[e]; }
    float inv = 1.f / sum;
    float4 o = {0.f, 0.f, 0.f, 0.f};
#pragma unroll
    for (int e = 0; e < EE; ++e) {
      float4 v4 = *(const float4*)(vp + ((size_t)b * EE + e) * DD + hh * HD + lane * 4);
      float p = sc[e] * inv;
      o.x += p * v4.x; o.y += p * v4.y; o.z += p * v4.z; o.w += p * v4.w;
    }
    float4 p4 = *(const float4*)(paged + qoff);
    float4 r;
    r.x = p4.x + ALPHA * o.x; r.y = p4.y + ALPHA * o.y;
    r.z = p4.z + ALPHA * o.z; r.w = p4.w + ALPHA * o.w;
    *(float4*)(h + qoff) = r;
  }
}

// ------------- LayerNorm forward (saves mean/rstd, bf16 output) ------------
__global__ __launch_bounds__(256) void k_ln_fwd(const float* __restrict__ h,
                                                const float* __restrict__ g,
                                                const float* __restrict__ bta,
                                                u16* __restrict__ hn,
                                                float* __restrict__ meanv,
                                                float* __restrict__ rstdv) {
  __shared__ float sh[8];
  int row = blockIdx.x;
  const float* x = h + (size_t)row * DD;
  float s = 0.f, ss = 0.f;
  for (int d = threadIdx.x; d < DD; d += 256) { float v = x[d]; s += v; ss += v * v; }
  float sm  = block_sum(s, sh);
  float sms = block_sum(ss, sh);
  float mean = sm * (1.f / DD);
  float var  = sms * (1.f / DD) - mean * mean;
  float rstd = rsqrtf(var + 1e-5f);
  if (threadIdx.x == 0) { meanv[row] = mean; rstdv[row] = rstd; }
  for (int d = threadIdx.x; d < DD; d += 256)
    hn[(size_t)row * DD + d] = f2bf((x[d] - mean) * rstd * g[d] + bta[d]);
}

// ------------- WMMA bf16 GEMM: C[M,N] = A[M,K] @ W[N,K]^T (+bias) ----------
// block = 256 threads (8 waves as 2x4), tile 64x256, BK=32; wave tile 32x64.
// TDM double-buffered pipeline: wave0 issues TENSOR_LOAD_TO_LDS for stage i+1
// while all waves run WMMA on stage i; s_wait_tensorcnt + split barrier sync.
#define GBM 64
#define GBN 256
#define GBK 32
__global__ __launch_bounds__(256) void k_gemm_bf16(const u16* __restrict__ A,
                                                   const u16* __restrict__ W,
                                                   const float* __restrict__ bias,
                                                   float* __restrict__ C,
                                                   int M, int N, int K) {
  __shared__ __align__(16) u16 As[2][GBM * GBK];
  __shared__ __align__(16) u16 Ws[2][GBN * GBK];
  int tid = threadIdx.x;
  int lane = tid & 31, wave = tid >> 5;
  int wm = wave >> 2;        // 0..1  (32-row band)
  int wn = wave & 3;         // 0..3  (64-col band)
  int m0 = blockIdx.y * GBM;
  int n0 = blockIdx.x * GBN;

  v8f acc[2][4];
#pragma unroll
  for (int i = 0; i < 2; ++i)
#pragma unroll
    for (int j = 0; j < 4; ++j) acc[i][j] = 0.f;

  int mlane = lane & 15, hi = lane >> 4;
  int nIter = K / GBK;

#if USE_TDM
  if (tid == 0) {   // TDM ignores EXEC; only wave0 reaches here (execz skip)
    tdm_load_2d((unsigned)(uintptr_t)&As[0][0], A + (size_t)m0 * K, K, GBM);
    tdm_load_2d((unsigned)(uintptr_t)&Ws[0][0], W + (size_t)n0 * K, K, GBN);
  }
#endif

  for (int it = 0; it < nIter; ++it) {
    int buf = it & 1;
    int k0 = it * GBK;
#if USE_TDM
    if (tid == 0) {
      if (it + 1 < nIter) {
        int k1 = k0 + GBK;
        tdm_load_2d((unsigned)(uintptr_t)&As[buf ^ 1][0],
                    A + (size_t)m0 * K + k1, K, GBM);
        tdm_load_2d((unsigned)(uintptr_t)&Ws[buf ^ 1][0],
                    W + (size_t)n0 * K + k1, K, GBN);
        __builtin_amdgcn_s_wait_tensorcnt(2);   // stage it complete (in-order)
      } else {
        __builtin_amdgcn_s_wait_tensorcnt(0);
      }
    }
    __syncthreads();                            // LDS stage `it` visible to all
#else
    {
      // manual staging fallback: A 64x32 (1 uint4/thread), W 256x32 (4/thread)
      *(uint4*)(As[buf] + (tid >> 2) * GBK + (tid & 3) * 8) =
          *(const uint4*)(A + (size_t)(m0 + (tid >> 2)) * K + k0 + (tid & 3) * 8);
      const uint4* wg = (const uint4*)(W + (size_t)(n0 + tid) * K + k0);
      uint4* wd = (uint4*)(Ws[buf] + tid * GBK);
      wd[0] = wg[0]; wd[1] = wg[1]; wd[2] = wg[2]; wd[3] = wg[3];
      __syncthreads();
    }
#endif

    // A fragments: 16x32 bf16; lane<16 -> K {0..7,16..23}, lane>=16 -> {8..15,24..31}
    FragBF a[2];
#pragma unroll
    for (int i = 0; i < 2; ++i) {
      int r = wm * 32 + i * 16 + mlane;
      a[i].u[0] = *(const uint4*)(As[buf] + r * GBK + 8 * hi);
      a[i].u[1] = *(const uint4*)(As[buf] + r * GBK + 16 + 8 * hi);
    }
    // B fragments: 32x16 (B[k,n]=W[n,k]); lane<16 -> K {0..7,16..23}, lane>=16 -> {8..15,24..31}
    FragBF bf[4];
#pragma unroll
    for (int j = 0; j < 4; ++j) {
      int r = wn * 64 + j * 16 + mlane;
      bf[j].u[0] = *(const uint4*)(Ws[buf] + r * GBK + 8 * hi);
      bf[j].u[1] = *(const uint4*)(Ws[buf] + r * GBK + 16 + 8 * hi);
    }
#pragma unroll
    for (int i = 0; i < 2; ++i)
#pragma unroll
      for (int j = 0; j < 4; ++j)
        acc[i][j] = __builtin_amdgcn_wmma_f32_16x16x32_bf16(
            false, a[i].v, false, bf[j].v, (short)0, acc[i][j], false, false);
    __syncthreads();   // done reading buf before TDM stage it+2 overwrites it
  }

  // epilogue: VGPR r -> row = r + 8*(lane>>4), col = lane&15
  int nlane = lane & 15, mbase = (lane >> 4) * 8;
#pragma unroll
  for (int i = 0; i < 2; ++i)
#pragma unroll
    for (int j = 0; j < 4; ++j) {
      int col = n0 + wn * 64 + j * 16 + nlane;
      float bv = bias ? bias[col] : 0.f;
#pragma unroll
      for (int r = 0; r < 8; ++r) {
        int row = m0 + wm * 32 + i * 16 + mbase + r;
        C[(size_t)row * N + col] = acc[i][j][r] + bv;
      }
    }
}

// ------------- GELU fwd / bwd (exact erf form) -----------------------------
__global__ __launch_bounds__(256) void k_gelu_fwd(const float* __restrict__ z,
                                                  u16* __restrict__ g1) {
  long long i = (long long)blockIdx.x * 256 + threadIdx.x;
  float x = z[i];
  g1[i] = f2bf(0.5f * x * (1.f + erff(x * 0.70710678118654752f)));
}

__global__ __launch_bounds__(256) void k_gelu_bwd(const float* __restrict__ dg,
                                                  const float* __restrict__ z,
                                                  u16* __restrict__ dz) {
  long long i = (long long)blockIdx.x * 256 + threadIdx.x;
  float x = z[i];
  float phi = 0.5f * (1.f + erff(x * 0.70710678118654752f));
  float pdf = 0.3989422804014327f * expf(-0.5f * x * x);
  dz[i] = f2bf(dg[i] * (phi + x * pdf));
}

// ------------- per-row ||ha|| and ha . kmean_n -----------------------------
__global__ __launch_bounds__(256) void k_rowstats(const float* __restrict__ ha,
                                                  const float* __restrict__ km,
                                                  float* __restrict__ rnorm,
                                                  float* __restrict__ rdot) {
  __shared__ float sh[8];
  int row = blockIdx.x;
  int b = row / SS;
  const float* x = ha + (size_t)row * DD;
  const float* kmb = km + (size_t)b * DD;
  float ss = 0.f, dt = 0.f;
  for (int d = threadIdx.x; d < DD; d += 256) {
    float v = x[d];
    ss += v * v;
    dt += v * kmb[d];
  }
  float tss = block_sum(ss, sh);
  float tdt = block_sum(dt, sh);
  if (threadIdx.x == 0) {
    rnorm[row] = fmaxf(sqrtf(tss), 1e-12f);
    rdot[row]  = tdt;
  }
}

// ------------- d_ha = -(1/N) * (kmean - (dot/n^2) ha) / n  (bf16) ----------
__global__ __launch_bounds__(256) void k_dha(const float* __restrict__ ha,
                                             const float* __restrict__ km,
                                             const float* __restrict__ rnorm,
                                             const float* __restrict__ rdot,
                                             u16* __restrict__ dha) {
  long long i = (long long)blockIdx.x * 256 + threadIdx.x;
  int row = (int)(i >> 12);          // /4096
  int d = (int)(i & 4095);
  int b = row >> 11;                 // /2048
  float invn = 1.f / rnorm[row];
  float coefA = -(1.f / (float)NROWS) * invn;
  float coefB = (1.f / (float)NROWS) * rdot[row] * invn * invn * invn;
  dha[i] = f2bf(coefA * km[(size_t)b * DD + d] + coefB * ha[i]);
}

// ------------- LN backward + fused SGD update of h -------------------------
__global__ __launch_bounds__(256) void k_ln_bwd(const float* __restrict__ dhn,
                                                const float* __restrict__ g,
                                                const float* __restrict__ meanv,
                                                const float* __restrict__ rstdv,
                                                float* __restrict__ h) {
  __shared__ float sh[8];
  int row = blockIdx.x;
  float mean = meanv[row], rstd = rstdv[row];
  const float* dy = dhn + (size_t)row * DD;
  float* x = h + (size_t)row * DD;
  float s1 = 0.f, s2 = 0.f;
  for (int d = threadIdx.x; d < DD; d += 256) {
    float dg = dy[d] * g[d];
    float xh = (x[d] - mean) * rstd;
    s1 += dg;
    s2 += dg * xh;
  }
  float m1 = block_sum(s1, sh) * (1.f / DD);
  float m2 = block_sum(s2, sh) * (1.f / DD);
  for (int d = threadIdx.x; d < DD; d += 256) {
    float dg = dy[d] * g[d];
    float xh = (x[d] - mean) * rstd;
    float dx = rstd * (dg - m1 - xh * m2);
    x[d] -= LRATE * dx;
  }
}

// ===========================================================================
extern "C" void kernel_launch(void* const* d_in, const int* in_sizes, int n_in,
                              void* d_out, int out_size, void* d_ws, size_t ws_size,
                              hipStream_t stream) {
  const float* paged = (const float*)d_in[0];
  const float* query = (const float*)d_in[1];
  const float* ek    = (const float*)d_in[2];
  const float* ev    = (const float*)d_in[3];
  const float* Wk    = (const float*)d_in[4];
  const float* Wv    = (const float*)d_in[5];
  const float* ln_g  = (const float*)d_in[6];
  const float* ln_b  = (const float*)d_in[7];
  const float* W1    = (const float*)d_in[8];
  const float* b1    = (const float*)d_in[9];
  const float* W2    = (const float*)d_in[10];
  const float* b2    = (const float*)d_in[11];
  const float* Wo    = (const float*)d_in[12];
  float* out = (float*)d_out;

  // ---- workspace carve-up (write-before-read everywhere) ----
  char* ws = (char*)d_ws;
  size_t off = 0;
  auto alloc = [&](size_t bytes) -> void* {
    void* p = ws + off;
    off += (bytes + 255) & ~(size_t)255;
    return p;
  };
  float* h      = (float*)alloc((size_t)NROWS * DD * 4);
  u16*   hnbf   = (u16*)  alloc((size_t)NROWS * DD * 2);   // reused for final h->bf16
  float* z1     = (float*)alloc((size_t)NROWS * DQ * 4);
  u16*   g1bf   = (u16*)  alloc((size_t)NROWS * DQ * 2);
  float* ha     = (float*)alloc((size_t)NROWS * DD * 4);   // reused as d_hn
  u16*   dhabf  = (u16*)  alloc((size_t)NROWS * DD * 2);
  float* dg1    = (float*)alloc((size_t)NROWS * DQ * 4);
  u16*   dz1bf  = (u16*)  alloc((size_t)NROWS * DQ * 2);
  float* kp     = (float*)alloc((size_t)BB * EE * DD * 4);
  float* vp     = (float*)alloc((size_t)BB * EE * DD * 4);
  float* km     = (float*)alloc((size_t)BB * DD * 4);
  float* meanv  = (float*)alloc((size_t)NROWS * 4);
  float* rstdv  = (float*)alloc((size_t)NROWS * 4);
  float* rnorm  = (float*)alloc((size_t)NROWS * 4);
  float* rdot   = (float*)alloc((size_t)NROWS * 4);
  u16*   w1b    = (u16*)  alloc((size_t)DQ * DD * 2);
  u16*   w1t    = (u16*)  alloc((size_t)DD * DQ * 2);
  u16*   w2b    = (u16*)  alloc((size_t)DD * DQ * 2);
  u16*   w2t    = (u16*)  alloc((size_t)DQ * DD * 2);
  u16*   wob    = (u16*)  alloc((size_t)DD * DD * 2);

  // ---- one-time weight conversions (re-run every call: deterministic) ----
  k_cvt_w<<<(DQ * DD) / 256, 256, 0, stream>>>(W1, w1b, w1t, DQ, DD);
  k_cvt_w<<<(DD * DQ) / 256, 256, 0, stream>>>(W2, w2b, w2t, DD, DQ);
  k_cvt_w<<<(DD * DD) / 256, 256, 0, stream>>>(Wo, wob, (u16*)nullptr, DD, DD);

  // ---- engram projections + kmean + attention fusion ----
  k_kvproj<<<DD, 256, 0, stream>>>(ek, Wk, kp);
  k_kvproj<<<DD, 256, 0, stream>>>(ev, Wv, vp);
  k_kmean<<<BB, 256, 0, stream>>>(ek, km);
  k_attn_fuse<<<NROWS, 256, 0, stream>>>(paged, query, kp, vp, h);

  // ---- TTA loop (tta_steps = 3 per reference config) ----
  dim3 gemm_nD(DD / GBN, NROWS / GBM);   // N=4096 outputs
  dim3 gemm_nQ(DQ / GBN, NROWS / GBM);   // N=1024 outputs
  for (int step = 0; step < TTA_STEPS; ++step) {
    k_ln_fwd<<<NROWS, 256, 0, stream>>>(h, ln_g, ln_b, hnbf, meanv, rstdv);
    // z1 = hn @ W1^T + b1
    k_gemm_bf16<<<gemm_nQ, 256, 0, stream>>>(hnbf, w1b, b1, z1, NROWS, DQ, DD);
    k_gelu_fwd<<<(NROWS * DQ) / 256, 256, 0, stream>>>(z1, g1bf);
    // ha = gelu(z1) @ W2^T + b2
    k_gemm_bf16<<<gemm_nD, 256, 0, stream>>>(g1bf, w2b, b2, ha, NROWS, DD, DQ);
    k_rowstats<<<NROWS, 256, 0, stream>>>(ha, km, rnorm, rdot);
    k_dha<<<(NROWS * DD) / 256, 256, 0, stream>>>(ha, km, rnorm, rdot, dhabf);
    // d_g1 = d_ha @ W2  (via W2^T stored as [DQ,DD])
    k_gemm_bf16<<<gemm_nQ, 256, 0, stream>>>(dhabf, w2t, (const float*)nullptr, dg1,
                                             NROWS, DQ, DD);
    k_gelu_bwd<<<(NROWS * DQ) / 256, 256, 0, stream>>>(dg1, z1, dz1bf);
    // d_hn = d_z1 @ W1 (via W1^T stored as [DD,DQ]); reuse `ha` as d_hn
    k_gemm_bf16<<<gemm_nD, 256, 0, stream>>>(dz1bf, w1t, (const float*)nullptr, ha,
                                             NROWS, DD, DQ);
    k_ln_bwd<<<NROWS, 256, 0, stream>>>(ha, ln_g, meanv, rstdv, h);
  }

  // ---- out = h @ Wo^T ----
  k_cvt_f32_bf16<<<(NROWS * DD) / 256, 256, 0, stream>>>(h, hnbf);
  k_gemm_bf16<<<gemm_nD, 256, 0, stream>>>(hnbf, wob, (const float*)nullptr, out,
                                           NROWS, DD, DD);
}
